// VectorizedSREKEnsemble_47605417508981
// MI455X (gfx1250) — matrix operand order repaired and checked by reading, not compiled, not run on hardware
//
#include <hip/hip_runtime.h>
#include <cmath>

// ---------------------------------------------------------------------------
// VectorizedSREKEnsemble on MI455X (gfx1250): bf16 WMMA 16x16x32, f32 accum.
// A=64 agents, B=4096 rows, D=50, H=192, O=3.
// Phase 1: convert fp32 weights -> padded bf16 in d_ws.
// Phase 2: one workgroup per (agent, 64-row tile); LDS ping-pong activations.
// Waves partition over N (3 n-tiles each) and sweep 4 M-tiles, so every
// B fragment held in registers is reused 4x -> 4x less L2 weight traffic.
// Epilogues: branchless GELU, hand-vectorized 2-wide (v_pk_* f32 ops) so the
// scalar v_exp/v_rcp TRANS chains interleave instead of stalling on v_nop.
// ---------------------------------------------------------------------------

typedef __attribute__((ext_vector_type(16))) __bf16 bf16x16;
typedef __attribute__((ext_vector_type(8)))  float  v8f;
typedef __attribute__((ext_vector_type(4)))  float  f32x4;
typedef __attribute__((ext_vector_type(2)))  float  f32x2;

#define A_N  64
#define B_N  4096
#define D_N  50
#define H_N  192
#define O_N  3
#define KP1  64          // D padded to multiple of 32
#define BT   64          // batch rows per workgroup
#define MT   4           // M-tiles (16 rows each) per block
#define NTW  3           // n-tiles per wave (4 waves x 3 = 12 = H/16)
#define LSTR 200         // LDS row stride in bf16 elems (192 + 8 pad, 16B aligned)
#define KS2  (H_N / 32)  // 6 K-steps for H=192
#define KS1  (KP1 / 32)  // 2 K-steps for padded D

union FragCast { f32x4 f4[2]; bf16x16 v; };

__device__ inline v8f wmma_bf16(bf16x16 a, bf16x16 b, v8f c) {
  // (neg_a, A, neg_b, B, c_mod, C, reuse_a, reuse_b)
  return __builtin_amdgcn_wmma_f32_16x16x32_bf16(false, a, false, b, (short)0, c,
                                                 false, false);
}

// B-fragment (32x16 bf16): lane n (0-15) holds K=0..15, lanes 16-31 hold
// K=16..31 -> each lane reads 16 contiguous bf16 of W row n at offset hi*16.
// Wrow already includes n*KP + hi*16.
__device__ inline bf16x16 load_b(const __bf16* __restrict__ Wrow, int kb) {
  FragCast u;
  const f32x4* p = reinterpret_cast<const f32x4*>(Wrow + kb);
  u.f4[0] = p[0];
  u.f4[1] = p[1];
  return u.v;
}

// A-fragment (16x32 bf16): lane m holds row m; lanes 0-15 K=0..15,
// lanes 16-31 K=16..31 -> two 16B LDS loads at kb+hi*16 and +8.
// lds_row already includes row*LSTR + hi*16.
__device__ inline bf16x16 load_a(const __bf16* lds_row, int kb) {
  FragCast u;
  u.f4[0] = *reinterpret_cast<const f32x4*>(lds_row + kb);
  u.f4[1] = *reinterpret_cast<const f32x4*>(lds_row + kb + 8);
  return u.v;
}

// Branchless tanh-form GELU, 2-wide. tanh(u) = 1 - 2/(1+exp(2u)) is sign-safe
// and stable at both extremes (exp->inf => rcp->0 => t->1; exp->0 => t->-1).
// Packed f32 VALU ops around two independent scalar exp/rcp TRANS chains.
__device__ inline f32x2 gelu2(f32x2 xv) {
  const f32x2 u = 0.7978845608028654f * xv * (0.044715f * xv * xv + 1.0f);
  const f32x2 w = 2.8853900817779268f * u;  // 2*log2(e)*u -> exp2(w)=exp(2u)
  f32x2 r;
  r.x = __builtin_amdgcn_rcpf(1.0f + __builtin_amdgcn_exp2f(w.x));
  r.y = __builtin_amdgcn_rcpf(1.0f + __builtin_amdgcn_exp2f(w.y));
  const f32x2 t = -2.0f * r + 1.0f;         // tanh(u)
  return xv * (0.5f * t + 0.5f);
}

// fp32 [G][RS][CS] -> bf16 [G][RD][CD], zero padded.
__global__ void srek_cvt(const float* __restrict__ src, __bf16* __restrict__ dst,
                         int G, int RS, int CS, int RD, int CD, long total) {
  long idx = (long)blockIdx.x * blockDim.x + threadIdx.x;
  if (idx >= total) return;
  int  c = (int)(idx % CD);
  long t = idx / CD;
  int  r = (int)(t % RD);
  int  g = (int)(t / RD);
  float v = 0.f;
  if (r < RS && c < CS) v = src[((size_t)g * RS + r) * CS + c];
  dst[idx] = (__bf16)v;
}

__global__ __launch_bounds__(128)
void srek_main(const float* __restrict__ x,
               const float* __restrict__ inw, const float* __restrict__ inb,
               const float* __restrict__ b1,  const float* __restrict__ b2,
               const float* __restrict__ b3,  const float* __restrict__ bo,
               const float* __restrict__ bs1, const float* __restrict__ bs2,
               const float* __restrict__ hnw, const float* __restrict__ hnb,
               const __bf16* __restrict__ wl1, const __bf16* __restrict__ ws1,
               const __bf16* __restrict__ wl2, const __bf16* __restrict__ ws2,
               const __bf16* __restrict__ wl3, const __bf16* __restrict__ wow,
               float* __restrict__ out) {
  __shared__ __bf16 sX[BT * LSTR];   // ping
  __shared__ __bf16 sY[BT * LSTR];   // pong

  const int a    = blockIdx.y;
  const int row0 = blockIdx.x * BT;
  const int tid  = threadIdx.x;
  const int lane = tid & 31;
  const int wv   = tid >> 5;         // 0..3
  const int ln   = lane & 15;
  const int hi   = lane >> 4;

  // ---- stage 0: x_norm = x*inw + inb -> sX[:, 0:64] (cols 50..63 = 0)
  for (int i = tid; i < BT * KP1; i += blockDim.x) {
    const int r = i >> 6, c = i & (KP1 - 1);
    float v = 0.f;
    if (c < D_N)
      v = x[(size_t)(row0 + r) * D_N + c] * inw[(size_t)a * D_N + c]
          + inb[(size_t)a * D_N + c];
    sX[r * LSTR + c] = (__bf16)v;
  }
  __syncthreads();

  // ---- layer1 + skip1: sX (K=64) -> sY = gelu(xn@W1^T+b1) + xn@S1^T+bs1
  {
    const __bf16* W = wl1 + (size_t)a * H_N * KP1;
    const __bf16* S = ws1 + (size_t)a * H_N * KP1;
#pragma unroll
    for (int nti = 0; nti < NTW; ++nti) {
      const int ng = (wv * NTW + nti) * 16 + ln;
      const __bf16* Wr = W + (size_t)ng * KP1 + hi * 16;
      const __bf16* Sr = S + (size_t)ng * KP1 + hi * 16;
      bf16x16 bw[KS1], bs[KS1];
#pragma unroll
      for (int k = 0; k < KS1; ++k) {
        bw[k] = load_b(Wr, k * 32);
        bs[k] = load_b(Sr, k * 32);
      }
      const float bb  = b1[(size_t)a * H_N + ng];
      const float bbs = bs1[(size_t)a * H_N + ng];
#pragma unroll
      for (int mt = 0; mt < MT; ++mt) {
        const __bf16* ar = sX + (mt * 16 + ln) * LSTR + hi * 16;
        v8f acc = {}, accs = {};
#pragma unroll
        for (int k = 0; k < KS1; ++k) {
          const bf16x16 af = load_a(ar, k * 32);
          acc  = wmma_bf16(af, bw[k], acc);
          accs = wmma_bf16(af, bs[k], accs);
        }
        float hv[8];
#pragma unroll
        for (int i = 0; i < 8; i += 2) {
          f32x2 v;  v.x  = acc[i]  + bb;   v.y  = acc[i + 1]  + bb;
          f32x2 sk; sk.x = accs[i] + bbs;  sk.y = accs[i + 1] + bbs;
          const f32x2 o = gelu2(v) + sk;
          hv[i] = o.x;  hv[i + 1] = o.y;
        }
        __bf16* dst = sY + (mt * 16 + hi * 8) * LSTR + ng;
#pragma unroll
        for (int i = 0; i < 8; ++i) dst[i * LSTR] = (__bf16)hv[i];
      }
    }
  }
  __syncthreads();

  // ---- layer2 + skip2: sY (K=192) -> sX
  {
    const __bf16* W = wl2 + (size_t)a * H_N * H_N;
    const __bf16* S = ws2 + (size_t)a * H_N * H_N;
#pragma unroll
    for (int nti = 0; nti < NTW; ++nti) {
      const int ng = (wv * NTW + nti) * 16 + ln;
      const __bf16* Wr = W + (size_t)ng * H_N + hi * 16;
      const __bf16* Sr = S + (size_t)ng * H_N + hi * 16;
      bf16x16 bw[KS2], bs[KS2];
#pragma unroll
      for (int k = 0; k < KS2; ++k) {
        bw[k] = load_b(Wr, k * 32);
        bs[k] = load_b(Sr, k * 32);
      }
      const float bb  = b2[(size_t)a * H_N + ng];
      const float bbs = bs2[(size_t)a * H_N + ng];
#pragma unroll
      for (int mt = 0; mt < MT; ++mt) {
        const __bf16* ar = sY + (mt * 16 + ln) * LSTR + hi * 16;
        v8f acc = {}, accs = {};
#pragma unroll
        for (int k = 0; k < KS2; ++k) {
          const bf16x16 af = load_a(ar, k * 32);
          acc  = wmma_bf16(af, bw[k], acc);
          accs = wmma_bf16(af, bs[k], accs);
        }
        float hv[8];
#pragma unroll
        for (int i = 0; i < 8; i += 2) {
          f32x2 v;  v.x  = acc[i]  + bb;   v.y  = acc[i + 1]  + bb;
          f32x2 sk; sk.x = accs[i] + bbs;  sk.y = accs[i + 1] + bbs;
          const f32x2 o = gelu2(v) + sk;
          hv[i] = o.x;  hv[i + 1] = o.y;
        }
        __bf16* dst = sX + (mt * 16 + hi * 8) * LSTR + ng;
#pragma unroll
        for (int i = 0; i < 8; ++i) dst[i * LSTR] = (__bf16)hv[i];
      }
    }
  }
  __syncthreads();

  // ---- layer3 + residual + hidden affine norm: sX (h2) -> sY (h3)
  {
    const __bf16* W = wl3 + (size_t)a * H_N * H_N;
#pragma unroll
    for (int nti = 0; nti < NTW; ++nti) {
      const int ng = (wv * NTW + nti) * 16 + ln;
      const __bf16* Wr = W + (size_t)ng * H_N + hi * 16;
      bf16x16 bw[KS2];
#pragma unroll
      for (int k = 0; k < KS2; ++k) bw[k] = load_b(Wr, k * 32);
      const float bb = b3[(size_t)a * H_N + ng];
      const float nw = hnw[(size_t)a * H_N + ng];
      const float nb = hnb[(size_t)a * H_N + ng];
#pragma unroll
      for (int mt = 0; mt < MT; ++mt) {
        const __bf16* ar = sX + (mt * 16 + ln) * LSTR + hi * 16;
        v8f acc = {};
#pragma unroll
        for (int k = 0; k < KS2; ++k)
          acc = wmma_bf16(load_a(ar, k * 32), bw[k], acc);
        // batch the residual loads into one clause, then 2-wide math
        const __bf16* src = sX + (mt * 16 + hi * 8) * LSTR + ng;
        float h2v[8];
#pragma unroll
        for (int i = 0; i < 8; ++i) h2v[i] = (float)src[i * LSTR];
        float hv[8];
#pragma unroll
        for (int i = 0; i < 8; i += 2) {
          f32x2 v;
          v.x = acc[i]     + bb + h2v[i];
          v.y = acc[i + 1] + bb + h2v[i + 1];
          const f32x2 o = v * nw + nb;
          hv[i] = o.x;  hv[i + 1] = o.y;
        }
        __bf16* dst = sY + (mt * 16 + hi * 8) * LSTR + ng;
#pragma unroll
        for (int i = 0; i < 8; ++i) dst[i * LSTR] = (__bf16)hv[i];
      }
    }
  }
  __syncthreads();

  // ---- output head: sY (K=192) x wow[A][16][192] (rows 3..15 zero).
  // Single n-tile globally -> back to per-wave M slice (wave w: rows 16w..).
  {
    const int m0 = wv * 16;
    const __bf16* ar = sY + (m0 + ln) * LSTR + hi * 16;
    const __bf16* Wr = wow + (size_t)a * 16 * H_N + (size_t)ln * H_N + hi * 16;
    v8f acc = {};
#pragma unroll
    for (int k = 0; k < KS2; ++k)
      acc = wmma_bf16(load_a(ar, k * 32), load_b(Wr, k * 32), acc);
    if (ln < O_N) {
      const float bb = bo[(size_t)a * O_N + ln];
#pragma unroll
      for (int i = 0; i < 8; ++i) {
        const int r = row0 + m0 + hi * 8 + i;
        out[((size_t)a * B_N + r) * O_N + ln] = acc[i] + bb;
      }
    }
  }
}

extern "C" void kernel_launch(void* const* d_in, const int* in_sizes, int n_in,
                              void* d_out, int out_size, void* d_ws, size_t ws_size,
                              hipStream_t stream) {
  (void)in_sizes; (void)n_in; (void)out_size; (void)ws_size;
  const float* x   = (const float*)d_in[0];
  const float* inw = (const float*)d_in[1];
  const float* inb = (const float*)d_in[2];
  const float* l1w = (const float*)d_in[3];
  const float* l1b = (const float*)d_in[4];
  const float* l2w = (const float*)d_in[5];
  const float* l2b = (const float*)d_in[6];
  const float* l3w = (const float*)d_in[7];
  const float* l3b = (const float*)d_in[8];
  const float* oww = (const float*)d_in[9];
  const float* owb = (const float*)d_in[10];
  const float* s1w = (const float*)d_in[11];
  const float* s1b = (const float*)d_in[12];
  const float* s2w = (const float*)d_in[13];
  const float* s2b = (const float*)d_in[14];
  const float* hnw = (const float*)d_in[15];
  const float* hnb = (const float*)d_in[16];
  float* out = (float*)d_out;

  // bf16 weight arena in d_ws (~17.7 MB)
  __bf16* ws = (__bf16*)d_ws;
  const size_t nL1 = (size_t)A_N * H_N * KP1;
  const size_t nH  = (size_t)A_N * H_N * H_N;
  __bf16* wl1 = ws;
  __bf16* ws1 = wl1 + nL1;
  __bf16* wl2 = ws1 + nL1;
  __bf16* ws2 = wl2 + nH;
  __bf16* wl3 = ws2 + nH;
  __bf16* wow = wl3 + nH;

  auto cvt = [&](const float* s, __bf16* d, int G, int RS, int CS, int RD, int CD) {
    long total = (long)G * RD * CD;
    int blocks = (int)((total + 255) / 256);
    srek_cvt<<<blocks, 256, 0, stream>>>(s, d, G, RS, CS, RD, CD, total);
  };
  cvt(l1w, wl1, A_N, H_N, D_N, H_N, KP1);
  cvt(s1w, ws1, A_N, H_N, D_N, H_N, KP1);
  cvt(l2w, wl2, A_N, H_N, H_N, H_N, H_N);
  cvt(s2w, ws2, A_N, H_N, H_N, H_N, H_N);
  cvt(l3w, wl3, A_N, H_N, H_N, H_N, H_N);
  cvt(oww, wow, A_N, O_N, H_N, 16, H_N);

  dim3 grid(B_N / BT, A_N);
  srek_main<<<grid, 128, 0, stream>>>(x, inw, inb, l1b, l2b, l3b, owb, s1b, s2b,
                                      hnw, hnb, wl1, ws1, wl2, ws2, wl3, wow, out);
}
